// Transformer_44229573214364
// MI455X (gfx1250) — matrix operand order repaired and checked by reading, not compiled, not run on hardware
//
#include <hip/hip_runtime.h>
#include <hip/hip_bf16.h>

// ---------------------------------------------------------------------------
// Two-layer attention "transformer" for MI455X (gfx1250, wave32, WMMA).
// Layer 1: full flash attention (bf16 WMMA, f32 accumulate, online softmax).
// Layer 2: only the last output row is needed -> reduced to one softmax-dot.
// Workspace requirement: ~64 MB (see layout in kernel_launch).
// ---------------------------------------------------------------------------

#define NROW 8192
#define DIM  384
#define KT   (DIM / 32)   // 12 K-steps of 32 for bf16 WMMA
#define DT   (DIM / 16)   // 24 output d-tiles of 16

typedef __attribute__((ext_vector_type(16))) __bf16 v16bf;
typedef __attribute__((ext_vector_type(8)))  __bf16 v8bf;
typedef __attribute__((ext_vector_type(8)))  float  v8f;

// gfx1250 async global->LDS path (ASYNCcnt-tracked), with safe fallback.
#if defined(__has_builtin)
#if __has_builtin(__builtin_amdgcn_global_load_async_to_lds_b128) && \
    __has_builtin(__builtin_amdgcn_s_wait_asynccnt)
#define USE_ASYNC_LDS 1
#endif
#endif
#ifndef USE_ASYNC_LDS
#define USE_ASYNC_LDS 0
#endif

#if USE_ASYNC_LDS
// Parameter types per the compiler diagnostic: b128 variant takes pointers to
// 16-byte int vectors; source in AS1 (global), destination in AS3 (LDS).
typedef int v4i_t __attribute__((vector_size(16)));
typedef __attribute__((address_space(1))) v4i_t gbl_v4i;
typedef __attribute__((address_space(3))) v4i_t lds_v4i;
__device__ __forceinline__ void async_copy_b128(const void* gsrc, void* ldst) {
  // Flat->AS1 is identity; flat LDS address [31:0] is the LDS byte offset
  // (aperture truncation rule), so integer round-trip casts are well-defined.
  __builtin_amdgcn_global_load_async_to_lds_b128(
      (gbl_v4i*)(uintptr_t)gsrc,
      (lds_v4i*)(uint32_t)(uintptr_t)ldst, 0, 0);
}
#endif

__device__ __forceinline__ v8f wmma_bf16(v16bf a, v16bf b, v8f c) {
  // (neg_a, A, neg_b, B, c_mod, C, reuse_a, reuse_b)
  return __builtin_amdgcn_wmma_f32_16x16x32_bf16(false, a, false, b, (short)0, c,
                                                 false, false);
}

// A-matrix 16x32 bf16 fragment (ISA 7.12.2): lane = (M = lane&15, half = lane>>4),
// element e -> vgpr r=e>>1; K = (half?8:0) + ((r&3)<<1) + (e&1) + ((r>>2)<<4).
__device__ __forceinline__ v16bf load_a_frag(const __bf16* base, int ld, int k0,
                                             int lane) {
  const int row = lane & 15;
  const int hb  = (lane >> 4) ? 8 : 0;
  v16bf a;
#pragma unroll
  for (int e = 0; e < 16; ++e) {
    const int r = e >> 1;
    const int k = hb + ((r & 3) << 1) + (e & 1) + ((r >> 2) << 4);
    a[e] = base[row * ld + k0 + k];
  }
  return a;
}

// B-matrix 32x16 bf16 fragment: lane = column (lane&15), half = lane>>4;
// element e -> K = half*8 + (e&7) + ((e>>3)<<4) (same K order as the A fragment).
// `base` is a K-contiguous layout: base[col * ld + k].
__device__ __forceinline__ v16bf load_b_frag(const __bf16* base, int ld, int k0,
                                             int lane) {
  const int col = lane & 15;
  const int kg  = (lane >> 4) << 3;
  v16bf b;
#pragma unroll
  for (int e = 0; e < 16; ++e) b[e] = base[col * ld + k0 + kg + (e & 7) + ((e >> 3) << 4)];
  return b;
}

// ---------------------------------------------------------------------------
// Kernel 0: f32 -> bf16 conversion of x and the shared weight matrices.
// ---------------------------------------------------------------------------
__global__ void convert_kernel(const float* __restrict__ x,
                               const float* __restrict__ Wq,
                               const float* __restrict__ Wk,
                               const float* __restrict__ Wv,
                               __bf16* __restrict__ xb,
                               __bf16* __restrict__ wqb,
                               __bf16* __restrict__ wkb,
                               __bf16* __restrict__ wvb) {
  const int stride = gridDim.x * blockDim.x;
  const int t0 = blockIdx.x * blockDim.x + threadIdx.x;
  for (int i = t0; i < NROW * DIM; i += stride) xb[i] = (__bf16)x[i];
  for (int i = t0; i < DIM * DIM; i += stride) {
    wqb[i] = (__bf16)Wq[i];
    wkb[i] = (__bf16)Wk[i];
    wvb[i] = (__bf16)Wv[i];
  }
}

// ---------------------------------------------------------------------------
// Kernel 1: projection GEMM  C[N,D] = A[N,D] @ W^T   (bf16 in, bf16 out,
// f32 accumulate).  W is row-major [D,D]; row d of W is the K-contiguous
// column d of W^T, so it feeds load_b_frag directly with no transpose.
// One wave per 16x16 output tile, 12 WMMAs along K.
// ---------------------------------------------------------------------------
__global__ __launch_bounds__(32) void proj_kernel(const __bf16* __restrict__ A,
                                                  const __bf16* __restrict__ Wb,
                                                  __bf16* __restrict__ C) {
  const int lane = threadIdx.x;
  const int i0 = blockIdx.x * 16;
  const int d0 = blockIdx.y * 16;
  v8f acc = {};
#pragma unroll
  for (int kt = 0; kt < KT; ++kt) {
    v16bf a = load_a_frag(A + i0 * DIM, DIM, kt * 32, lane);
    v16bf b = load_b_frag(Wb + d0 * DIM, DIM, kt * 32, lane);
    acc = wmma_bf16(a, b, acc);
  }
  const int half = lane >> 4;
  const int n    = lane & 15;
#pragma unroll
  for (int r = 0; r < 8; ++r) {
    const int m = half * 8 + r;
    C[(i0 + m) * DIM + d0 + n] = (__bf16)acc[r];
  }
}

// ---------------------------------------------------------------------------
// Kernel 2: flash attention, layer 1.  8 waves per block; each wave owns a
// 16-row Q tile (fragments resident) and a full 16x384 f32 accumulator.
// Per 32-column step:
//   - K tile staged into LDS via GLOBAL_LOAD_ASYNC_TO_LDS_B128 (ASYNCcnt),
//   - V tile loaded as b128 and transposed into LDS (K-contiguous for P@V),
//   - S = Q K^T (24 WMMA), online softmax (shfl_xor row reductions across the
//     16-lane half that matches the C layout), P staged through LDS into
//     A-fragments, acc += P V (24 WMMA).
// ---------------------------------------------------------------------------
__global__ __launch_bounds__(256) void attn_kernel(const __bf16* __restrict__ Qb,
                                                   const __bf16* __restrict__ Kb,
                                                   const __bf16* __restrict__ Vb,
                                                   float* __restrict__ h1,
                                                   __bf16* __restrict__ h1b) {
  __shared__ __bf16 Kc[32 * DIM];       // [local col c][k=d]  (K-contiguous)
  __shared__ __bf16 Vt[DIM * 32];       // [d][local row r]    (K-contiguous in r)
  __shared__ __bf16 Pl[8][16 * 32];     // per-wave P relayout buffer

  const int tid  = threadIdx.x;
  const int lane = tid & 31;
  const int w    = tid >> 5;
  const int i0   = blockIdx.x * (8 * 16) + w * 16;

  // Resident Q fragments for this wave's 16 rows.
  v16bf qf[KT];
#pragma unroll
  for (int kt = 0; kt < KT; ++kt)
    qf[kt] = load_a_frag(Qb + i0 * DIM, DIM, kt * 32, lane);

  v8f zero = {};
  v8f acc[DT];
#pragma unroll
  for (int t = 0; t < DT; ++t) acc[t] = zero;

  float mrow[8], lrow[8];
#pragma unroll
  for (int r = 0; r < 8; ++r) { mrow[r] = -3.0e38f; lrow[r] = 0.0f; }

  const int half = lane >> 4;
  const int ncol = lane & 15;

  for (int j0 = 0; j0 < NROW; j0 += 32) {
    __syncthreads();   // previous tile fully consumed by all waves

    // --- Stage K tile (32 x DIM bf16 = 24576 B = 1536 x 16B chunks). ---
#if USE_ASYNC_LDS
    {
      const char* gsrc = (const char*)(Kb + (size_t)j0 * DIM);
      char* ldst = (char*)&Kc[0];
#pragma unroll
      for (int i = 0; i < 6; ++i) {
        const int chunk = tid + i * 256;
        async_copy_b128(gsrc + chunk * 16, ldst + chunk * 16);
      }
    }
#else
    {
      const v8bf* gsrc = (const v8bf*)(Kb + (size_t)j0 * DIM);
      v8bf* ldst = (v8bf*)&Kc[0];
#pragma unroll
      for (int i = 0; i < 6; ++i) {
        const int chunk = tid + i * 256;
        ldst[chunk] = gsrc[chunk];
      }
    }
#endif

    // --- Stage V tile: b128 global reads, 2-byte transposed LDS scatter. ---
#pragma unroll
    for (int i = 0; i < 6; ++i) {
      const int t  = tid + i * 256;          // 1536 chunks: 32 rows x 48 chunks
      const int c  = t / 48;                 // local row (j)
      const int d8 = (t - c * 48) * 8;       // starting d of this chunk
      const v8bf vv = *(const v8bf*)(Vb + (size_t)(j0 + c) * DIM + d8);
#pragma unroll
      for (int e = 0; e < 8; ++e) Vt[(d8 + e) * 32 + c] = vv[e];
    }

    if (j0 + 32 < NROW && tid < 32)
      __builtin_prefetch(Kb + (size_t)(j0 + 32 + tid) * DIM, 0, 1);

#if USE_ASYNC_LDS
    __builtin_amdgcn_s_wait_asynccnt(0);
#endif
    __syncthreads();

    // S tiles for columns [j0, j0+16) and [j0+16, j0+32).
    v8f s0 = zero, s1 = zero;
#pragma unroll
    for (int kt = 0; kt < KT; ++kt) {
      v16bf b0 = load_b_frag(Kc, DIM, kt * 32, lane);
      v16bf b1 = load_b_frag(Kc + 16 * DIM, DIM, kt * 32, lane);
      s0 = wmma_bf16(qf[kt], b0, s0);
      s1 = wmma_bf16(qf[kt], b1, s1);
    }

    // Online softmax across the 32 new columns.
    float scale[8];
#pragma unroll
    for (int r = 0; r < 8; ++r) {
      float v = fmaxf(s0[r], s1[r]);
#pragma unroll
      for (int off = 1; off < 16; off <<= 1)
        v = fmaxf(v, __shfl_xor(v, off, 32));
      const float mnew = fmaxf(mrow[r], v);
      scale[r] = __expf(mrow[r] - mnew);
      mrow[r]  = mnew;
      const float p0 = __expf(s0[r] - mnew);
      const float p1 = __expf(s1[r] - mnew);
      float rs = p0 + p1;
#pragma unroll
      for (int off = 1; off < 16; off <<= 1) rs += __shfl_xor(rs, off, 32);
      lrow[r] = lrow[r] * scale[r] + rs;
      const int m = half * 8 + r;
      Pl[w][m * 32 + ncol]      = (__bf16)p0;
      Pl[w][m * 32 + 16 + ncol] = (__bf16)p1;
    }

    // Rescale accumulator (row m = r + 8*half matches scale[r] per half-wave).
#pragma unroll
    for (int t = 0; t < DT; ++t) {
#pragma unroll
      for (int r = 0; r < 8; ++r) acc[t][r] *= scale[r];
    }

    // P (16x32) back as an A-fragment; acc += P @ V-block.
    v16bf pa = load_a_frag(&Pl[w][0], 32, 0, lane);
#pragma unroll
    for (int t = 0; t < DT; ++t) {
      v16bf bv = load_b_frag(Vt + (t * 16) * 32, 32, 0, lane);
      acc[t] = wmma_bf16(pa, bv, acc[t]);
    }
  }

  // Epilogue: normalize and store h1 (f32 for reference fidelity, bf16 feed).
  float inv[8];
#pragma unroll
  for (int r = 0; r < 8; ++r) inv[r] = 1.0f / lrow[r];
#pragma unroll
  for (int t = 0; t < DT; ++t) {
#pragma unroll
    for (int r = 0; r < 8; ++r) {
      const int m = half * 8 + r;
      const float val = acc[t][r] * inv[r];
      h1[(i0 + m) * DIM + t * 16 + ncol]  = val;
      h1b[(i0 + m) * DIM + t * 16 + ncol] = (__bf16)val;
    }
  }
}

// ---------------------------------------------------------------------------
// Kernel 3: layer-2 collapse.  Only row N-1 of the second attention matters:
//   s = softmax(q2 . K2[j,:]),  z = s @ V2,  out = sigmoid(sum(ffws * z)).
// Single workgroup, block-wide reductions through LDS.
// ---------------------------------------------------------------------------
__global__ __launch_bounds__(1024) void final_kernel(
    const __bf16* __restrict__ Q2b, const __bf16* __restrict__ K2b,
    const __bf16* __restrict__ V2b, const float* __restrict__ ffws,
    float* __restrict__ out, float* __restrict__ sbuf) {
  __shared__ float red[1024];
  __shared__ __bf16 q2s[DIM];
  const int tid = threadIdx.x;
  const int nth = blockDim.x;

  for (int k = tid; k < DIM; k += nth) q2s[k] = Q2b[(NROW - 1) * DIM + k];
  __syncthreads();

  // Raw scores + block max.
  float lmax = -3.0e38f;
  for (int j = tid; j < NROW; j += nth) {
    float acc = 0.0f;
    const __bf16* kr = K2b + (size_t)j * DIM;
#pragma unroll 4
    for (int k = 0; k < DIM; ++k) acc += (float)q2s[k] * (float)kr[k];
    sbuf[j] = acc;
    lmax = fmaxf(lmax, acc);
  }
  red[tid] = lmax;
  __syncthreads();
  for (int s = nth >> 1; s > 0; s >>= 1) {
    if (tid < s) red[tid] = fmaxf(red[tid], red[tid + s]);
    __syncthreads();
  }
  const float gmax = red[0];
  __syncthreads();

  // exp + block sum.
  float lsum = 0.0f;
  for (int j = tid; j < NROW; j += nth) {
    const float e = __expf(sbuf[j] - gmax);
    sbuf[j] = e;
    lsum += e;
  }
  red[tid] = lsum;
  __syncthreads();
  for (int s = nth >> 1; s > 0; s >>= 1) {
    if (tid < s) red[tid] += red[tid + s];
    __syncthreads();
  }
  const float invsum = 1.0f / red[0];
  __syncthreads();

  // z_d = sum_j s_j * V2[j,d]; fold ffws in; block sum; sigmoid.
  float part = 0.0f;
  if (tid < DIM) {
    float z = 0.0f;
    for (int j = 0; j < NROW; ++j) z += sbuf[j] * (float)V2b[(size_t)j * DIM + tid];
    part = ffws[tid] * z * invsum;
  }
  red[tid] = part;
  __syncthreads();
  for (int s = nth >> 1; s > 0; s >>= 1) {
    if (tid < s) red[tid] += red[tid + s];
    __syncthreads();
  }
  if (tid == 0) out[0] = 1.0f / (1.0f + __expf(-red[0]));
}

// ---------------------------------------------------------------------------
extern "C" void kernel_launch(void* const* d_in, const int* in_sizes, int n_in,
                              void* d_out, int out_size, void* d_ws,
                              size_t ws_size, hipStream_t stream) {
  const float* x    = (const float*)d_in[0];
  const float* Wq   = (const float*)d_in[1];
  const float* Wk   = (const float*)d_in[2];
  const float* Wv   = (const float*)d_in[3];
  const float* ffws = (const float*)d_in[4];
  float* out        = (float*)d_out;

  // Workspace carve-up (~64 MB total).
  size_t off = 0;
  auto take = [&](size_t bytes) -> void* {
    void* p = (char*)d_ws + off;
    off += (bytes + 255) & ~(size_t)255;
    return p;
  };
  const size_t ND2 = (size_t)NROW * DIM * sizeof(__bf16);
  const size_t DD2 = (size_t)DIM * DIM * sizeof(__bf16);
  __bf16* xb   = (__bf16*)take(ND2);
  __bf16* wqb  = (__bf16*)take(DD2);
  __bf16* wkb  = (__bf16*)take(DD2);
  __bf16* wvb  = (__bf16*)take(DD2);
  __bf16* Qb   = (__bf16*)take(ND2);
  __bf16* Kb   = (__bf16*)take(ND2);
  __bf16* Vb   = (__bf16*)take(ND2);
  float*  h1   = (float*)take((size_t)NROW * DIM * sizeof(float));
  __bf16* h1b  = (__bf16*)take(ND2);
  __bf16* Q2b  = (__bf16*)take(ND2);
  __bf16* K2b  = (__bf16*)take(ND2);
  __bf16* V2b  = (__bf16*)take(ND2);
  float*  sbuf = (float*)take((size_t)NROW * sizeof(float));
  (void)ws_size; (void)in_sizes; (void)n_in; (void)out_size;

  // 0) precision conversion
  convert_kernel<<<256, 256, 0, stream>>>(x, Wq, Wk, Wv, xb, wqb, wkb, wvb);

  // 1) layer-1 projections (bf16 WMMA GEMMs)
  dim3 pg(NROW / 16, DIM / 16);
  proj_kernel<<<pg, 32, 0, stream>>>(xb, wqb, Qb);
  proj_kernel<<<pg, 32, 0, stream>>>(xb, wkb, Kb);
  proj_kernel<<<pg, 32, 0, stream>>>(xb, wvb, Vb);

  // 2) layer-1 flash attention -> h1
  attn_kernel<<<NROW / 128, 256, 0, stream>>>(Qb, Kb, Vb, h1, h1b);

  // 3) layer-2 projections on h1
  proj_kernel<<<pg, 32, 0, stream>>>(h1b, wqb, Q2b);
  proj_kernel<<<pg, 32, 0, stream>>>(h1b, wkb, K2b);
  proj_kernel<<<pg, 32, 0, stream>>>(h1b, wvb, V2b);

  // 4) layer-2 last-row attention + readout
  final_kernel<<<1, 1024, 0, stream>>>(Q2b, K2b, V2b, ffws, out, sbuf);
}
